// ConcatBiInteraction_22874995819093
// MI455X (gfx1250) — matrix operand order repaired and checked by reading, not compiled, not run on hardware
//
#include <hip/hip_runtime.h>
#include <hip/hip_bf16.h>

typedef __attribute__((ext_vector_type(16))) __bf16 v16bf;
typedef __attribute__((ext_vector_type(8)))  float  v8f;

// ---------------- WMMA fragment helpers (ISA 7.12.2 layouts) ----------------
// 16-bit A matrix 16x32 (MxK): lanes 0-15 -> M, VGPR v holds K pair
//   k = (v&3)*2 + (v>>2)*16 + (lane>=16 ? 8 : 0)
__device__ __forceinline__ v16bf load_a_frag_f32(const float* __restrict__ src,
                                                 int row0, int lda, int k0, int lane) {
  int m = lane & 15, half = lane >> 4;
  const float* r = src + (row0 + m) * lda;
  v16bf a;
#pragma unroll
  for (int v = 0; v < 8; ++v) {
    int k = k0 + ((v & 3) << 1) + ((v >> 2) << 4) + (half << 3);
    a[2 * v]     = (__bf16)r[k];
    a[2 * v + 1] = (__bf16)r[k + 1];
  }
  return a;
}

__device__ __forceinline__ v16bf load_a_frag_lds(const __bf16* __restrict__ src,
                                                 int row0, int lda, int k0, int lane) {
  int m = lane & 15, half = lane >> 4;
  const __bf16* r = src + (row0 + m) * lda;
  v16bf a;
#pragma unroll
  for (int v = 0; v < 8; ++v) {
    int k = k0 + ((v & 3) << 1) + ((v >> 2) << 4) + (half << 3);
    a[2 * v]     = r[k];
    a[2 * v + 1] = r[k + 1];
  }
  return a;
}

// 16-bit B matrix 32x16 (KxN): same K order as A,
//   k = (v&3)*2 + (v>>2)*16 + (lane>=16 ? 8 : 0)
__device__ __forceinline__ v16bf load_b_frag_f32(const float* __restrict__ w,
                                                 int k0, int ldb, int n0, int lane) {
  int n = lane & 15, half = lane >> 4;
  v16bf b;
#pragma unroll
  for (int v = 0; v < 8; ++v) {
    int k = k0 + ((v & 3) << 1) + ((v >> 2) << 4) + (half << 3);
    b[2 * v]     = (__bf16)w[k * ldb + n0 + n];
    b[2 * v + 1] = (__bf16)w[(k + 1) * ldb + n0 + n];
  }
  return b;
}

// ---------------- K1: atom_part[1024,128] = atom_embed @ att1_W[128:,:] -----
__global__ void atom_proj_kernel(const float* __restrict__ atom_embed,
                                 const float* __restrict__ att1_W,
                                 float* __restrict__ atom_part) {
  int tid = threadIdx.x, lane = tid & 31, wave = tid >> 5;
  int gw = blockIdx.x * 4 + wave;       // 0..511 tiles
  int mt = gw >> 3, nt = gw & 7;        // 64 M-tiles x 8 N-tiles
  const float* wbot = att1_W + 128 * 128;
  v8f c = {};
#pragma unroll
  for (int ks = 0; ks < 128; ks += 32) {
    v16bf a  = load_a_frag_f32(atom_embed, mt * 16, 128, ks, lane);
    v16bf bb = load_b_frag_f32(wbot, ks, 128, nt * 16, lane);
    c = __builtin_amdgcn_wmma_f32_16x16x32_bf16(false, a, false, bb, (short)0, c, false, false);
  }
  int col = nt * 16 + (lane & 15), half = lane >> 4;
#pragma unroll
  for (int r = 0; r < 8; ++r)
    atom_part[(mt * 16 + r + 8 * half) * 128 + col] = c[r];
}

// ---------------- K2: fused scores per (segment, 64-l chunk) ----------------
#define PH_LD 129   // pad: (ll*129+j)%64 spreads banks across l
#define AP_LD 132
__global__ void score_kernel(const float* __restrict__ protSeq,
                             const float* __restrict__ att1_W,
                             const float* __restrict__ att1_b,
                             const float* __restrict__ att2_W,
                             const float* __restrict__ att2_b,
                             const float* __restrict__ atom_part,
                             float* __restrict__ Wpart,   // [1024][8]
                             float* __restrict__ Wp) {    // [32][512]
  __shared__ float ph[64 * PH_LD];     // protein projection chunk
  __shared__ float apS[32 * AP_LD];    // atom projections for this segment
  __shared__ float biasS[128];
  __shared__ float w2S[128];
  __shared__ float Wbuf[32 * 64];
  int tid = threadIdx.x, lane = tid & 31, wave = tid >> 5;
  int b = blockIdx.x >> 3, ch = blockIdx.x & 7, l0 = ch * 64;

  for (int i = tid; i < 32 * 128; i += 256) {
    int at = i >> 7, j = i & 127;
    apS[at * AP_LD + j] = atom_part[(b * 32 + at) * 128 + j];
  }
  if (tid < 128) { biasS[tid] = att1_b[tid]; w2S[tid] = att2_W[tid]; }

  // phase 1: prot_hid[l0:l0+64, 0:128] via WMMA into LDS (32 tiles / 8 waves)
  const float* X = protSeq + (b * 512 + l0) * 128;
#pragma unroll
  for (int i = 0; i < 4; ++i) {
    int t = wave + 8 * i;              // 0..31
    int mt = t >> 3, nt = t & 7;
    v8f c = {};
#pragma unroll
    for (int ks = 0; ks < 128; ks += 32) {
      v16bf a  = load_a_frag_f32(X, mt * 16, 128, ks, lane);
      v16bf bb = load_b_frag_f32(att1_W, ks, 128, nt * 16, lane);
      c = __builtin_amdgcn_wmma_f32_16x16x32_bf16(false, a, false, bb, (short)0, c, false, false);
    }
    int col = nt * 16 + (lane & 15), half = lane >> 4;
#pragma unroll
    for (int r = 0; r < 8; ++r)
      ph[(mt * 16 + r + 8 * half) * PH_LD + col] = c[r];
  }
  __syncthreads();

  // phase 2: scores W[n,l] = 5*tanh(sum_j tanh(ph+ap+b1)*w2 + b2)
  int ll = tid & 63, ag = tid >> 6;
  float b2 = att2_b[0];
#pragma unroll
  for (int ai = 0; ai < 8; ++ai) {
    int at = ag * 8 + ai;
    const float* phr = &ph[ll * PH_LD];
    const float* apr = &apS[at * AP_LD];
    float s = 0.f;
    for (int j = 0; j < 128; ++j)
      s += tanhf(phr[j] + apr[j] + biasS[j]) * w2S[j];
    Wbuf[at * 64 + ll] = 5.0f * tanhf(s + b2);
  }
  __syncthreads();

  if (tid < 64) {                       // Wp[b,l]: max over the 32 atoms
    float m = -1e30f;
    for (int at = 0; at < 32; ++at) m = fmaxf(m, Wbuf[at * 64 + tid]);
    Wp[b * 512 + l0 + tid] = m;
  } else if (tid < 96) {                // per-atom chunk max
    int at = tid - 64;
    float m = -1e30f;
    for (int l = 0; l < 64; ++l) m = fmaxf(m, Wbuf[at * 64 + l]);
    Wpart[(b * 32 + at) * 8 + ch] = m;
  }
}

// ---------------- K3: Wc[n] = exp(max_l W[n,l]) -----------------------------
__global__ void atom_max_kernel(const float* __restrict__ Wpart,
                                float* __restrict__ Wc) {
  int n = blockIdx.x * 256 + threadIdx.x;
  float m = -1e30f;
#pragma unroll
  for (int c = 0; c < 8; ++c) m = fmaxf(m, Wpart[n * 8 + c]);
  Wc[n] = expf(m);
}

// ---------------- K4: per-segment pooling -> z[32,256] ----------------------
__global__ void pool_kernel(const float* __restrict__ Wc,
                            const float* __restrict__ atom_embed,
                            const float* __restrict__ Wp,
                            const float* __restrict__ protSeq,
                            float* __restrict__ z) {
  __shared__ float aaS[32];
  __shared__ float red[128];
  __shared__ float apS[512];
  int b = blockIdx.x, t = threadIdx.x;   // 128 threads
  if (t < 32) aaS[t] = Wc[b * 32 + t];
  __syncthreads();
  if (t == 0) { float s = 0.f; for (int i = 0; i < 32; ++i) s += aaS[i]; red[0] = s; }
  __syncthreads();
  float Sc = red[0];
  __syncthreads();

  float apool = 0.f;
  for (int i = 0; i < 32; ++i)
    apool += (aaS[i] / Sc) * atom_embed[(b * 32 + i) * 128 + t];

  // softmax over Wp[b, :]
  float v[4]; float lm = -1e30f;
#pragma unroll
  for (int q = 0; q < 4; ++q) { v[q] = Wp[b * 512 + q * 128 + t]; lm = fmaxf(lm, v[q]); }
  red[t] = lm; __syncthreads();
  for (int off = 64; off > 0; off >>= 1) {
    if (t < off) red[t] = fmaxf(red[t], red[t + off]);
    __syncthreads();
  }
  float M = red[0]; __syncthreads();
  float ls = 0.f;
#pragma unroll
  for (int q = 0; q < 4; ++q) { v[q] = expf(v[q] - M); ls += v[q]; }
  red[t] = ls; __syncthreads();
  for (int off = 64; off > 0; off >>= 1) {
    if (t < off) red[t] += red[t + off];
    __syncthreads();
  }
  float S = red[0]; __syncthreads();
#pragma unroll
  for (int q = 0; q < 4; ++q) apS[q * 128 + t] = v[q] / S;
  __syncthreads();

  float pp = 0.f;
  for (int l = 0; l < 512; ++l)
    pp += apS[l] * protSeq[(b * 512 + l) * 128 + t];

  z[b * 256 + t] = apool;
  z[b * 256 + 128 + t] = pp;
}

// ---------------- K5: fused MLP (WMMA) -> out[32] ---------------------------
__global__ void mlp_kernel(const float* __restrict__ z,
                           const float* __restrict__ d1_W, const float* __restrict__ d1_b,
                           const float* __restrict__ d2_W, const float* __restrict__ d2_b,
                           const float* __restrict__ out_W, const float* __restrict__ out_b,
                           float* __restrict__ out) {
  __shared__ __bf16 h1S[32 * 512];
  __shared__ __bf16 h2S[32 * 256];
  int tid = threadIdx.x, lane = tid & 31, wave = tid >> 5, half = lane >> 4;

  // layer 1: (32x256)@(256x512), relu
  for (int t = wave; t < 64; t += 8) {
    int mt = t & 1, nt = t >> 1;
    v8f c = {};
#pragma unroll
    for (int ks = 0; ks < 256; ks += 32) {
      v16bf a  = load_a_frag_f32(z, mt * 16, 256, ks, lane);
      v16bf bb = load_b_frag_f32(d1_W, ks, 512, nt * 16, lane);
      c = __builtin_amdgcn_wmma_f32_16x16x32_bf16(false, a, false, bb, (short)0, c, false, false);
    }
    int col = nt * 16 + (lane & 15);
    float bias = d1_b[col];
#pragma unroll
    for (int r = 0; r < 8; ++r)
      h1S[(mt * 16 + r + 8 * half) * 512 + col] = (__bf16)fmaxf(c[r] + bias, 0.f);
  }
  __syncthreads();

  // layer 2: (32x512)@(512x256), relu
  for (int t = wave; t < 32; t += 8) {
    int mt = t & 1, nt = t >> 1;
    v8f c = {};
#pragma unroll
    for (int ks = 0; ks < 512; ks += 32) {
      v16bf a  = load_a_frag_lds(h1S, mt * 16, 512, ks, lane);
      v16bf bb = load_b_frag_f32(d2_W, ks, 256, nt * 16, lane);
      c = __builtin_amdgcn_wmma_f32_16x16x32_bf16(false, a, false, bb, (short)0, c, false, false);
    }
    int col = nt * 16 + (lane & 15);
    float bias = d2_b[col];
#pragma unroll
    for (int r = 0; r < 8; ++r)
      h2S[(mt * 16 + r + 8 * half) * 256 + col] = (__bf16)fmaxf(c[r] + bias, 0.f);
  }
  __syncthreads();

  if (tid < 32) {
    float s = 0.f;
    for (int j = 0; j < 256; ++j)
      s += (float)h2S[tid * 256 + j] * out_W[j];
    out[tid] = s + out_b[0];
  }
}

// ---------------- host launcher --------------------------------------------
extern "C" void kernel_launch(void* const* d_in, const int* in_sizes, int n_in,
                              void* d_out, int out_size, void* d_ws, size_t ws_size,
                              hipStream_t stream) {
  const float* atom_embed = (const float*)d_in[0];
  const float* protSeq    = (const float*)d_in[1];
  // d_in[2] atom_splits: atoms are 32 contiguous per segment (n >> 5) -> unused
  const float* att1_W = (const float*)d_in[3];
  const float* att1_b = (const float*)d_in[4];
  const float* att2_W = (const float*)d_in[5];
  const float* att2_b = (const float*)d_in[6];
  const float* d1_W   = (const float*)d_in[7];
  const float* d1_b   = (const float*)d_in[8];
  const float* d2_W   = (const float*)d_in[9];
  const float* d2_b   = (const float*)d_in[10];
  const float* out_W  = (const float*)d_in[11];
  const float* out_b  = (const float*)d_in[12];
  float* out = (float*)d_out;

  char* ws = (char*)d_ws;
  float* atom_part = (float*)(ws);                                  // 1024*128
  float* Wpart     = (float*)(ws + 524288);                         // 1024*8
  float* Wp        = (float*)(ws + 524288 + 32768);                 // 32*512
  float* Wc        = (float*)(ws + 524288 + 32768 + 65536);         // 1024
  float* z         = (float*)(ws + 524288 + 32768 + 65536 + 4096);  // 32*256

  atom_proj_kernel<<<128, 128, 0, stream>>>(atom_embed, att1_W, atom_part);
  score_kernel<<<256, 256, 0, stream>>>(protSeq, att1_W, att1_b, att2_W, att2_b,
                                        atom_part, Wpart, Wp);
  atom_max_kernel<<<4, 256, 0, stream>>>(Wpart, Wc);
  pool_kernel<<<32, 128, 0, stream>>>(Wc, atom_embed, Wp, protSeq, z);
  mlp_kernel<<<1, 256, 0, stream>>>(z, d1_W, d1_b, d2_W, d2_b, out_W, out_b, out);
}
